// ResGATBlock_15006615734302
// MI455X (gfx1250) — compile-verified, matrix operand order checked by reading, not run on hardware
//
#include <hip/hip_runtime.h>
#include <hip/hip_fp16.h>

// ---------------------------------------------------------------------------
// ResGATBlock for MI455X (gfx1250, wave32).
//   N=50000 nodes, E=800000 edges, D=128, H=8 heads x C=16.
// GEMMs run on v_wmma_f32_16x16x32_f16 with register-resident f16 weights;
// edge phase is L2-resident gather/scatter (node arrays ~80MB << 192MB L2).
// ---------------------------------------------------------------------------

#define NN 50000
#define EE 800000
#define DD 128
#define HH 8
#define CC 16
#define NEG_SLOPE 0.15f
#define GEMM_GRID 512

typedef __attribute__((ext_vector_type(16))) _Float16 v16h;
typedef __attribute__((ext_vector_type(8)))  float    v8f;

// Monotonic float <-> uint mapping for atomicMax-based segment max.
__device__ __forceinline__ unsigned fkey(float f) {
  unsigned u = __float_as_uint(f);
  return (u & 0x80000000u) ? ~u : (u | 0x80000000u);
}
__device__ __forceinline__ float fdec(unsigned k) {
  unsigned u = (k & 0x80000000u) ? (k ^ 0x80000000u) : ~k;
  return __uint_as_float(u);
}

// ---------------------------------------------------------------------------
// 0) One-time weight pack: f32 row-major [K][N] -> f16 column-major [N][K].
//    Makes every B-fragment chunk contiguous (b128 loads, zero in-loop cvts).
// ---------------------------------------------------------------------------
__global__ void pack_weights_kernel(const float* __restrict__ Wl,
                                    const float* __restrict__ Wr,
                                    _Float16* __restrict__ WlT,
                                    _Float16* __restrict__ WrT) {
  const int t = blockIdx.x * blockDim.x + threadIdx.x;   // DD*DD = 64*256
  const int k = t >> 7;
  const int n = t & (DD - 1);
  WlT[n * DD + k] = (_Float16)Wl[t];
  WrT[n * DD + k] = (_Float16)Wr[t];
}

// ---------------------------------------------------------------------------
// 1) LayerNorm + downcast to f16. One wave (32 lanes) per node, 4 floats/lane.
// ---------------------------------------------------------------------------
__global__ void ln_to_f16_kernel(const float* __restrict__ x,
                                 const float* __restrict__ g,
                                 const float* __restrict__ b,
                                 _Float16* __restrict__ h16) {
  const int wave = threadIdx.x >> 5;
  const int lane = threadIdx.x & 31;
  const int node = blockIdx.x * 8 + wave;          // NN = 6250*8 exactly
  const float4 v = ((const float4*)(x + (size_t)node * DD))[lane];

  float s = v.x + v.y + v.z + v.w;
#pragma unroll
  for (int m = 1; m < 32; m <<= 1) s += __shfl_xor(s, m, 32);
  const float mu = s * (1.0f / DD);

  const float d0 = v.x - mu, d1 = v.y - mu, d2 = v.z - mu, d3 = v.w - mu;
  float q = d0 * d0 + d1 * d1 + d2 * d2 + d3 * d3;
#pragma unroll
  for (int m = 1; m < 32; m <<= 1) q += __shfl_xor(q, m, 32);
  const float inv = rsqrtf(q * (1.0f / DD) + 1e-5f);

  const int f = lane * 4;
  _Float16* o = h16 + (size_t)node * DD + f;
  o[0] = (_Float16)(d0 * inv * g[f + 0] + b[f + 0]);
  o[1] = (_Float16)(d1 * inv * g[f + 1] + b[f + 1]);
  o[2] = (_Float16)(d2 * inv * g[f + 2] + b[f + 2]);
  o[3] = (_Float16)(d3 * inv * g[f + 3] + b[f + 3]);
}

// ---------------------------------------------------------------------------
// 2) Dual GEMM via WMMA: xl = h@W_l + b_l, xr = h@W_r + b_r.
//    Block = 8 waves; wave w owns output columns [16w, 16w+16) and keeps its
//    128x16 f16 B-slices of BOTH weight matrices in 64 VGPRs, then
//    grid-strides over 16-row M tiles (NN = 3125*16 exactly).
//    Steady-state loop: 8x global_load_b128 (A) -> 8x v_wmma -> 16 stores.
//    ISA 7.12.2 fragment layout:
//      16-bit A 16x32: lanes 0-15 -> M=lane, K base 0; lanes 16-31 -> K base 8.
//        v16h element j -> K = base + j (j<8) | 8 + base + j (j>=8)
//      B 32x16 mirrors A with N = lane&15.
//      f32 C/D 16x16: VGPR r -> M = r + 8*(lane>>4), N = lane&15.
// ---------------------------------------------------------------------------
__global__ void gemm_dual_wmma_kernel(const _Float16* __restrict__ h16,
                                      const _Float16* __restrict__ WlT,
                                      const _Float16* __restrict__ WrT,
                                      const float* __restrict__ bl,
                                      const float* __restrict__ br,
                                      float* __restrict__ xl,
                                      float* __restrict__ xr) {
  const int wave = threadIdx.x >> 5;
  const int lane = threadIdx.x & 31;
  const int colBase = wave * 16;
  const int half = lane >> 4;
  const int mn   = lane & 15;       // M for A-frag, N for B/C/D frags
  const int kb   = half * 8;
  const int col  = colBase + mn;

  // Register-resident B fragments for all 4 K-steps, both matrices.
  v16h BL[4], BR[4];
  {
    const _Float16* wl = WlT + (size_t)col * DD;
    const _Float16* wr = WrT + (size_t)col * DD;
#pragma unroll
    for (int s = 0; s < 4; ++s) {
#pragma unroll
      for (int j = 0; j < 16; ++j) {
        const int k = 32 * s + kb + j + ((j >= 8) ? 8 : 0);
        BL[s][j] = wl[k];
        BR[s][j] = wr[k];
      }
    }
  }
  const float cbl = bl[col];
  const float cbr = br[col];

  for (int mt = blockIdx.x; mt < NN / 16; mt += GEMM_GRID) {
    const int row0 = mt * 16;
    const _Float16* arow = h16 + (size_t)(row0 + mn) * DD;

    v8f accL = {};
    v8f accR = {};
#pragma unroll
    for (int s = 0; s < 4; ++s) {
      v16h a;
#pragma unroll
      for (int j = 0; j < 16; ++j) {
        const int k = 32 * s + kb + j + ((j >= 8) ? 8 : 0);
        a[j] = arow[k];
      }
      accL = __builtin_amdgcn_wmma_f32_16x16x32_f16(false, a, false, BL[s],
                                                    (short)0, accL, false, false);
      accR = __builtin_amdgcn_wmma_f32_16x16x32_f16(false, a, false, BR[s],
                                                    (short)0, accR, false, false);
    }

#pragma unroll
    for (int r = 0; r < 8; ++r) {
      const int row = row0 + r + 8 * half;
      xl[(size_t)row * DD + col] = accL[r] + cbl;
      xr[(size_t)row * DD + col] = accR[r] + cbr;
    }
  }
}

// ---------------------------------------------------------------------------
// 3) Edge attention logits + segment max. One wave per edge; lane handles 4
//    consecutive features (head = lane>>2, 4 lanes per head).
// ---------------------------------------------------------------------------
__global__ void edge_alpha_kernel(const float* __restrict__ xl,
                                  const float* __restrict__ xr,
                                  const int* __restrict__ src,
                                  const int* __restrict__ dst,
                                  const float* __restrict__ att,
                                  float* __restrict__ alpha,
                                  unsigned* __restrict__ mkey) {
  const int wave = threadIdx.x >> 5;
  const int lane = threadIdx.x & 31;
  const int e = blockIdx.x * 8 + wave;             // EE = 100000*8 exactly
  const int s = src[e];
  const int d = dst[e];

  const float4 a4 = ((const float4*)(xl + (size_t)s * DD))[lane];
  const float4 b4 = ((const float4*)(xr + (size_t)d * DD))[lane];
  const float4 w4 = ((const float4*)att)[lane];

  float t0 = a4.x + b4.x; t0 = t0 > 0.f ? t0 : NEG_SLOPE * t0;
  float t1 = a4.y + b4.y; t1 = t1 > 0.f ? t1 : NEG_SLOPE * t1;
  float t2 = a4.z + b4.z; t2 = t2 > 0.f ? t2 : NEG_SLOPE * t2;
  float t3 = a4.w + b4.w; t3 = t3 > 0.f ? t3 : NEG_SLOPE * t3;
  float v = t0 * w4.x + t1 * w4.y + t2 * w4.z + t3 * w4.w;

  v += __shfl_xor(v, 1, 4);
  v += __shfl_xor(v, 2, 4);

  if ((lane & 3) == 0) {
    const int h = lane >> 2;
    alpha[(size_t)e * HH + h] = v;
    atomicMax(&mkey[(size_t)d * HH + h], fkey(v));
  }
}

// ---------------------------------------------------------------------------
// 4) ex = exp(alpha - max) in place, accumulate softmax denominators.
// ---------------------------------------------------------------------------
__global__ void exp_denom_kernel(float* __restrict__ alpha,
                                 const int* __restrict__ dst,
                                 const unsigned* __restrict__ mkey,
                                 float* __restrict__ denom) {
  const int t = blockIdx.x * blockDim.x + threadIdx.x;  // EE*HH = 25000*256
  const int e = t >> 3;
  const int h = t & 7;
  const int d = dst[e];
  const float ex = __expf(alpha[t] - fdec(mkey[(size_t)d * HH + h]));
  alpha[t] = ex;
  atomicAdd(&denom[(size_t)d * HH + h], ex);
}

// ---------------------------------------------------------------------------
// 5) Weighted scatter-add of messages. Thread handles 4 features of one edge.
// ---------------------------------------------------------------------------
__global__ void aggregate_kernel(const float* __restrict__ xl,
                                 const float* __restrict__ alpha,
                                 const float* __restrict__ denom,
                                 const int* __restrict__ src,
                                 const int* __restrict__ dst,
                                 float* __restrict__ agg) {
  const int t = blockIdx.x * blockDim.x + threadIdx.x;  // EE*32 = 100000*256
  const int e = t >> 5;
  const int q = t & 31;
  const int s = src[e];
  const int d = dst[e];
  const int f = q * 4;
  const int h = f >> 4;

  const float a = alpha[(size_t)e * HH + h] /
                  (denom[(size_t)d * HH + h] + 1e-16f);
  const float4 xv = ((const float4*)(xl + (size_t)s * DD))[q];
  float* o = agg + (size_t)d * DD + f;
  atomicAdd(o + 0, xv.x * a);
  atomicAdd(o + 1, xv.y * a);
  atomicAdd(o + 2, xv.z * a);
  atomicAdd(o + 3, xv.w * a);
}

// ---------------------------------------------------------------------------
// 6) out = x + relu(agg + bias)
// ---------------------------------------------------------------------------
__global__ void finalize_kernel(const float* __restrict__ x,
                                const float* __restrict__ agg,
                                const float* __restrict__ bias,
                                float* __restrict__ out) {
  const int t = blockIdx.x * blockDim.x + threadIdx.x;  // NN*DD = 25000*256
  const float v = agg[t] + bias[t & (DD - 1)];
  out[t] = x[t] + (v > 0.f ? v : 0.f);
}

// ---------------------------------------------------------------------------
extern "C" void kernel_launch(void* const* d_in, const int* in_sizes, int n_in,
                              void* d_out, int out_size, void* d_ws, size_t ws_size,
                              hipStream_t stream) {
  (void)in_sizes; (void)n_in; (void)out_size; (void)ws_size;

  const float* x    = (const float*)d_in[0];
  const int*   ei   = (const int*)  d_in[1];
  const float* ln_g = (const float*)d_in[2];
  const float* ln_b = (const float*)d_in[3];
  const float* W_l  = (const float*)d_in[4];
  const float* b_l  = (const float*)d_in[5];
  const float* W_r  = (const float*)d_in[6];
  const float* b_r  = (const float*)d_in[7];
  const float* att  = (const float*)d_in[8];
  const float* bias = (const float*)d_in[9];
  const int* src = ei;
  const int* dst = ei + EE;

  // Workspace layout (~118 MB), 256B aligned slices.
  char* ws = (char*)d_ws;
  size_t off = 0;
  auto take = [&](size_t bytes) -> char* {
    char* p = ws + off;
    off = (off + bytes + 255) & ~(size_t)255;
    return p;
  };
  _Float16* h16  = (_Float16*)take((size_t)NN * DD * sizeof(_Float16));
  _Float16* WlT  = (_Float16*)take((size_t)DD * DD * sizeof(_Float16));
  _Float16* WrT  = (_Float16*)take((size_t)DD * DD * sizeof(_Float16));
  float*    xl   = (float*)   take((size_t)NN * DD * sizeof(float));
  float*    xr   = (float*)   take((size_t)NN * DD * sizeof(float));
  float*    alpha= (float*)   take((size_t)EE * HH * sizeof(float));
  unsigned* mkey = (unsigned*)take((size_t)NN * HH * sizeof(unsigned));
  float*    denom= (float*)   take((size_t)NN * HH * sizeof(float));
  float*    agg  = (float*)   take((size_t)NN * DD * sizeof(float));

  // Zero accumulators (graph-capture safe). mkey==0 is the minimal ordered key.
  hipMemsetAsync(mkey,  0, (size_t)NN * HH * sizeof(unsigned), stream);
  hipMemsetAsync(denom, 0, (size_t)NN * HH * sizeof(float),    stream);
  hipMemsetAsync(agg,   0, (size_t)NN * DD * sizeof(float),    stream);

  pack_weights_kernel<<<(DD * DD) / 256, 256, 0, stream>>>(W_l, W_r, WlT, WrT);
  ln_to_f16_kernel   <<<NN / 8,          256, 0, stream>>>(x, ln_g, ln_b, h16);
  gemm_dual_wmma_kernel<<<GEMM_GRID,     256, 0, stream>>>(h16, WlT, WrT, b_l, b_r, xl, xr);
  edge_alpha_kernel  <<<EE / 8,          256, 0, stream>>>(xl, xr, src, dst, att, alpha, mkey);
  exp_denom_kernel   <<<(EE * HH) / 256, 256, 0, stream>>>(alpha, dst, mkey, denom);
  aggregate_kernel   <<<(EE * 32) / 256, 256, 0, stream>>>(xl, alpha, denom, src, dst, agg);
  finalize_kernel    <<<(NN * DD) / 256, 256, 0, stream>>>(x, agg, bias, (float*)d_out);
}
